// UniAttention_84456236908628
// MI455X (gfx1250) — matrix operand order repaired and checked by reading, not compiled, not run on hardware
//
#include <hip/hip_runtime.h>

// ---------------------------------------------------------------------------
// Fused masked cross-attention (ESIM "UniAttention") for MI455X / gfx1250.
// B=16, P=2048, H=2048, D=1024, fp32 in/out, bf16 WMMA internally.
//
// Kernel 1 (pre-pass): tiled convert+transpose of hypothesis into d_ws:
//    hypB [b][h][d] bf16 (64 MB)  -- K=d contiguous for S-GEMM B-frags
//    hypT [b][d][h] bf16 (64 MB)  -- K=h contiguous for O-GEMM B-frags
//    (requires ws_size >= 128 MB)
// Kernel 2 (main): 256 threads = 8 wave32 = 2 M-chunks x 4 K/D-slices.
//    Block owns 32 premise rows of one batch; flash-style online softmax.
//    Hyp tiles are staged with GLOBAL_LOAD_ASYNC_TO_LDS_B128 (ASYNCcnt),
//    bypassing VGPRs entirely (CDNA5 async global->LDS path).
// ---------------------------------------------------------------------------

typedef __attribute__((ext_vector_type(16))) __bf16 v16bf;
typedef __attribute__((ext_vector_type(8)))  float  v8f;

#define WMMA_BF16(A, Bm, C) \
    __builtin_amdgcn_wmma_f32_16x16x32_bf16(false, (A), false, (Bm), (short)0, (C), false, false)

namespace {
constexpr int Bn = 16;    // batch
constexpr int Pn = 2048;  // premise len
constexpr int Hn = 2048;  // hypothesis len
constexpr int Dn = 1024;  // feature dim
constexpr int NT = 32;    // hypothesis tile rows per iteration (K of 2nd GEMM)
constexpr int MC = 2;     // M-chunks (16 rows each) per block
constexpr int NWS = 4;    // K/D-slice waves per M-chunk

// Padded LDS strides (elements), keeping 16B alignment for b128 frag loads.
constexpr int PSTR = Dn + 8;  // premise row stride (bf16)
constexpr int HSTR = Dn + 8;  // hyp row stride (bf16)
constexpr int TSTR = NT + 8;  // hypT per-d stride (bf16): 80B, 16B aligned
constexpr int SSTR = NT;      // spart row stride (f32)
constexpr int WSTR = NT + 8;  // wbuf row stride (bf16)

constexpr size_t SM_PREM  = (size_t)MC * 16 * PSTR * 2;        // 66048
constexpr size_t SM_HYP   = (size_t)NT * HSTR * 2;             // 66048
constexpr size_t SM_HYPT  = (size_t)Dn * TSTR * 2;             // 81920
constexpr size_t SM_SPART = (size_t)MC * NWS * 16 * SSTR * 4;  // 16384
constexpr size_t SM_WBUF  = (size_t)MC * NWS * 16 * WSTR * 2;  // 10240
constexpr size_t SMEM_BYTES = SM_PREM + SM_HYP + SM_HYPT + SM_SPART + SM_WBUF;
}  // namespace

__device__ __forceinline__ unsigned short f2bf(float f) {
    unsigned u = __builtin_bit_cast(unsigned, f);
    unsigned r = u + 0x7FFFu + ((u >> 16) & 1u);  // round-to-nearest-even
    return (unsigned short)(r >> 16);
}

__device__ __forceinline__ void store4bf(unsigned short* p, float4 v) {
    unsigned lo = (unsigned)f2bf(v.x) | ((unsigned)f2bf(v.y) << 16);
    unsigned hi = (unsigned)f2bf(v.z) | ((unsigned)f2bf(v.w) << 16);
    uint2 u;  u.x = lo;  u.y = hi;
    *(uint2*)p = u;
}

struct FragBits { uint4 lo; uint4 hi; };  // 32 bytes = 8 VGPRs

__device__ __forceinline__ v16bf ldfrag(const unsigned short* plo,
                                        const unsigned short* phi) {
    FragBits f;
    f.lo = *(const uint4*)plo;  // ds_load_b128
    f.hi = *(const uint4*)phi;  // ds_load_b128
    return __builtin_bit_cast(v16bf, f);
}

// CDNA5 async global->LDS copy (16B per lane), tracked on ASYNCcnt.
// VDST = wave-relative LDS byte address, VADDR = 64-bit global address (GV).
__device__ __forceinline__ void async_copy_b128(const void* lds_dst,
                                                const void* global_src) {
    unsigned int lds_addr =
        (unsigned int)(unsigned long long)(uintptr_t)lds_dst;  // low 32 = LDS off
    unsigned long long gaddr = (unsigned long long)(uintptr_t)global_src;
    asm volatile("global_load_async_to_lds_b128 %0, %1, off"
                 :: "v"(lds_addr), "v"(gaddr)
                 : "memory");
}

__device__ __forceinline__ void wait_asynccnt0() {
#if __has_builtin(__builtin_amdgcn_s_wait_asynccnt)
    __builtin_amdgcn_s_wait_asynccnt(0);
#else
    asm volatile("s_wait_asynccnt 0x0" ::: "memory");
#endif
}

// ---------------------------------------------------------------------------
// Pre-pass: 64x64 tiles, f32 -> bf16 row-major + bf16 transposed.
// grid = (D/64, H/64, B), block = 256.
// ---------------------------------------------------------------------------
__global__ __launch_bounds__(256, 4)
void hyp_convert_transpose_kernel(const float* __restrict__ hyp_g,
                                  unsigned short* __restrict__ hypB_g,
                                  unsigned short* __restrict__ hypT_g) {
    __shared__ unsigned short tile[64][72];  // padded vs bank conflicts

    const int tid = threadIdx.x;
    const int b   = blockIdx.z;
    const int h0  = blockIdx.y * 64;
    const int d0  = blockIdx.x * 64;

    // Phase 1: load f32, convert, write row-major bf16 + stash in LDS.
    for (int i = tid; i < 64 * 64 / 4; i += 256) {   // 1024 float4 chunks
        int row = i >> 4;            // h within tile
        int c4  = (i & 15) << 2;     // d within tile
        float4 v = *(const float4*)(hyp_g + ((size_t)b * Hn + h0 + row) * Dn + d0 + c4);
        unsigned short b0 = f2bf(v.x), b1 = f2bf(v.y), b2 = f2bf(v.z), b3 = f2bf(v.w);
        uint2 u;
        u.x = (unsigned)b0 | ((unsigned)b1 << 16);
        u.y = (unsigned)b2 | ((unsigned)b3 << 16);
        *(uint2*)(hypB_g + ((size_t)b * Hn + h0 + row) * Dn + d0 + c4) = u;
        tile[row][c4 + 0] = b0;
        tile[row][c4 + 1] = b1;
        tile[row][c4 + 2] = b2;
        tile[row][c4 + 3] = b3;
    }
    __syncthreads();

    // Phase 2: write transposed bf16 [b][d][h], coalesced over h.
    for (int i = tid; i < 64 * 64 / 4; i += 256) {
        int dr = i >> 4;             // d within tile
        int h4 = (i & 15) << 2;      // h within tile
        uint2 u;
        u.x = (unsigned)tile[h4 + 0][dr] | ((unsigned)tile[h4 + 1][dr] << 16);
        u.y = (unsigned)tile[h4 + 2][dr] | ((unsigned)tile[h4 + 3][dr] << 16);
        *(uint2*)(hypT_g + ((size_t)b * Dn + d0 + dr) * Hn + h0 + h4) = u;
    }
}

// ---------------------------------------------------------------------------
// Main fused attention kernel.  grid = (P/32, B), block = 256 (8 wave32).
// Wave wv: mc = wv>>2 selects 16-row M-chunk, ws = wv&3 selects the
// 256-wide K-slice (S-GEMM) / D-slice (O-GEMM).
// ---------------------------------------------------------------------------
__global__ __launch_bounds__(256, 1)
void uniattn_fused_kernel(const float* __restrict__ prem_g,
                          const float* __restrict__ pmask_g,
                          const unsigned short* __restrict__ hypB_g,
                          const unsigned short* __restrict__ hypT_g,
                          const float* __restrict__ hmask_g,
                          float* __restrict__ out_g) {
    extern __shared__ char smem[];
    unsigned short* premL = (unsigned short*)smem;
    unsigned short* hypL  = (unsigned short*)(smem + SM_PREM);
    unsigned short* hypTl = (unsigned short*)(smem + SM_PREM + SM_HYP);
    float*          spart = (float*)        (smem + SM_PREM + SM_HYP + SM_HYPT);
    unsigned short* wbuf  = (unsigned short*)(smem + SM_PREM + SM_HYP + SM_HYPT + SM_SPART);

    const int tid    = threadIdx.x;
    const int lane   = tid & 31;
    const int wv     = tid >> 5;       // 0..7
    const int mc     = wv >> 2;        // 0..1  M-chunk
    const int ws     = wv & 3;         // 0..3  K/D slice
    const int lane16 = lane & 15;
    const int half   = lane >> 4;
    const int klo    = half ? 8 : 0;   // A-frag K sub-offset per ISA layout

    const int b  = blockIdx.y;
    const int p0 = blockIdx.x * (MC * 16);

    // ---- Prologue: premise tile (32 x 1024) f32 -> bf16 LDS --------------
    for (int i = tid; i < MC * 16 * Dn / 4; i += 256) {
        int row = i >> 8;
        int d4  = (i & 255) << 2;
        float4 v = *(const float4*)(prem_g + ((size_t)b * Pn + p0 + row) * Dn + d4);
        store4bf(premL + row * PSTR + d4, v);
    }

    v8f o[16];
#pragma unroll
    for (int t = 0; t < 16; ++t) {
#pragma unroll
        for (int j = 0; j < 8; ++j) o[t][j] = 0.0f;
    }
    float mrow[8], lrow[8];
#pragma unroll
    for (int r = 0; r < 8; ++r) { mrow[r] = -3.0e38f; lrow[r] = 0.0f; }

    constexpr float kScale = 0.03125f;  // 1/sqrt(1024)

    // ---- Main loop over hypothesis tiles ---------------------------------
    for (int h0 = 0; h0 < Hn; h0 += NT) {
        __syncthreads();  // previous iteration done with hyp LDS

        // Row-major bf16 tile (32 x 1024): async global->LDS, no VGPR staging.
        for (int i = tid; i < NT * Dn / 8; i += 256) {     // 4096 x 16B
            int row = i >> 7;
            int c8  = (i & 127) << 3;
            async_copy_b128(hypL + row * HSTR + c8,
                            hypB_g + ((size_t)b * Hn + h0 + row) * Dn + c8);
        }
        // Transposed bf16 tile (1024 d x 32 h): async global->LDS.
        for (int i = tid; i < Dn * NT / 8; i += 256) {     // 4096 x 16B
            int d   = i >> 2;
            int seg = (i & 3) << 3;
            async_copy_b128(hypTl + d * TSTR + seg,
                            hypT_g + ((size_t)b * Dn + d) * Hn + h0 + seg);
        }
        // Prefetch next row-major tile (global_prefetch_b8).
        if (h0 + NT < Hn) {
            __builtin_prefetch(hypB_g + ((size_t)b * Hn + h0 + NT + (tid & 31)) * Dn
                               + (size_t)(tid >> 5) * 128, 0, 1);
        }
        wait_asynccnt0();   // our async copies have landed in LDS
        __syncthreads();    // ... and everyone else's too

        // ---- S = Prem * Hyp^T : K-slice [ws*256, ws*256+256) -------------
        v8f s0, s1;
#pragma unroll
        for (int j = 0; j < 8; ++j) { s0[j] = 0.0f; s1[j] = 0.0f; }
        const unsigned short* premRow = premL + (mc * 16 + lane16) * PSTR;
#pragma unroll
        for (int kk = 0; kk < 8; ++kk) {
            const int k0 = ws * 256 + kk * 32;
            v16bf a = ldfrag(premRow + k0 + half * 16, premRow + k0 + half * 16 + 8);
            const unsigned short* b0p = hypL + lane16 * HSTR + k0 + half * 16;
            const unsigned short* b1p = hypL + (lane16 + 16) * HSTR + k0 + half * 16;
            v16bf bb0 = ldfrag(b0p, b0p + 8);
            v16bf bb1 = ldfrag(b1p, b1p + 8);
            s0 = WMMA_BF16(a, bb0, s0);
            s1 = WMMA_BF16(a, bb1, s1);
        }
        {
            float* sp = spart + (mc * NWS + ws) * (16 * SSTR);
#pragma unroll
            for (int r = 0; r < 8; ++r) {
                int m = r + 8 * half;
                sp[m * SSTR + lane16]      = s0[r];
                sp[m * SSTR + 16 + lane16] = s1[r];
            }
        }
        __syncthreads();

        // ---- Reduce K-partials + online masked softmax (per M-chunk, ----
        // ---- replicated across the 4 slice-waves of that chunk) ----------
        float km0 = hmask_g[(size_t)b * Hn + h0 + lane16];
        float km1 = hmask_g[(size_t)b * Hn + h0 + 16 + lane16];
        v8f p0v, p1v;
        float alpha[8];
#pragma unroll
        for (int r = 0; r < 8; ++r) {
            int m = r + 8 * half;
            float a0 = 0.0f, a1 = 0.0f;
#pragma unroll
            for (int w = 0; w < NWS; ++w) {
                const float* sp = spart + (mc * NWS + w) * (16 * SSTR) + m * SSTR;
                a0 += sp[lane16];
                a1 += sp[16 + lane16];
            }
            // reference: sim*kmask, softmax, *kmask, renormalize
            float x0 = a0 * kScale * km0;
            float x1 = a1 * kScale * km1;
            float mx = fmaxf(x0, x1);
#pragma unroll
            for (int off = 1; off <= 8; off <<= 1)
                mx = fmaxf(mx, __shfl_xor(mx, off, 32));
            float mnew = fmaxf(mrow[r], mx);
            float al   = __expf(mrow[r] - mnew);
            float e0   = __expf(x0 - mnew) * km0;
            float e1   = __expf(x1 - mnew) * km1;
            float rs   = e0 + e1;
#pragma unroll
            for (int off = 1; off <= 8; off <<= 1)
                rs += __shfl_xor(rs, off, 32);
            lrow[r]  = lrow[r] * al + rs;
            mrow[r]  = mnew;
            alpha[r] = al;
            p0v[r]   = e0;
            p1v[r]   = e1;
        }
        // Rescale O accumulators by alpha.
#pragma unroll
        for (int t = 0; t < 16; ++t) {
#pragma unroll
            for (int r = 0; r < 8; ++r) o[t][r] *= alpha[r];
        }
        // W (C-layout) -> per-wave LDS -> A-layout fragment (DS in-order
        // within a wave makes the cross-lane store->load safe).
        {
            unsigned short* wb = wbuf + wv * (16 * WSTR);
#pragma unroll
            for (int r = 0; r < 8; ++r) {
                int m = r + 8 * half;
                wb[m * WSTR + lane16]      = f2bf(p0v[r]);
                wb[m * WSTR + 16 + lane16] = f2bf(p1v[r]);
            }
            v16bf wf = ldfrag(wb + lane16 * WSTR + half * 16,
                              wb + lane16 * WSTR + half * 16 + 8);

            // ---- O += W * Hyp : D-slice [ws*256, ws*256+256) -------------
#pragma unroll
            for (int t = 0; t < 16; ++t) {
                const unsigned short* bp =
                    hypTl + (ws * 256 + t * 16 + lane16) * TSTR + half * 16;
                v16bf bt = ldfrag(bp, bp + 8);
                o[t] = WMMA_BF16(wf, bt, o[t]);
            }
        }
    }

    // ---- Epilogue: normalize, apply premise mask, store f32 --------------
    float sc[8];
#pragma unroll
    for (int r = 0; r < 8; ++r) {
        int m = mc * 16 + r + 8 * half;
        float pm = pmask_g[(size_t)b * Pn + p0 + m];
        sc[r] = pm / (lrow[r] + 1e-13f);
    }
#pragma unroll
    for (int t = 0; t < 16; ++t) {
        int dcol = ws * 256 + t * 16 + lane16;
#pragma unroll
        for (int r = 0; r < 8; ++r) {
            int m = mc * 16 + r + 8 * half;
            out_g[((size_t)b * Pn + p0 + m) * Dn + dcol] = o[t][r] * sc[r];
        }
    }
}

extern "C" void kernel_launch(void* const* d_in, const int* in_sizes, int n_in,
                              void* d_out, int out_size, void* d_ws, size_t ws_size,
                              hipStream_t stream) {
    (void)in_sizes; (void)n_in; (void)out_size; (void)ws_size;
    const float* prem  = (const float*)d_in[0];
    const float* pmask = (const float*)d_in[1];
    const float* hyp   = (const float*)d_in[2];
    const float* hmask = (const float*)d_in[3];
    float* out = (float*)d_out;

    // Workspace layout: hypB bf16 (64 MB) | hypT bf16 (64 MB).
    unsigned short* hypB_ws = (unsigned short*)d_ws;
    unsigned short* hypT_ws = hypB_ws + (size_t)Bn * Hn * Dn;

    // Pre-pass: convert + transpose hypothesis once.
    {
        dim3 grid(Dn / 64, Hn / 64, Bn);
        hyp_convert_transpose_kernel<<<grid, 256, 0, stream>>>(hyp, hypB_ws, hypT_ws);
    }

    (void)hipFuncSetAttribute((const void*)uniattn_fused_kernel,
                              hipFuncAttributeMaxDynamicSharedMemorySize,
                              (int)SMEM_BYTES);
    {
        dim3 grid(Pn / (MC * 16), Bn);
        uniattn_fused_kernel<<<grid, 256, SMEM_BYTES, stream>>>(
            prem, pmask, hypB_ws, hypT_ws, hmask, out);
    }
}
